// acsasrec_decode2_35261681500929
// MI455X (gfx1250) — matrix operand order repaired and checked by reading, not compiled, not run on hardware
//
#include <hip/hip_runtime.h>
#include <cstdint>

// ---------------------------------------------------------------------------
// CDNA5 (gfx1250) wave32 WMMA bf16 transformer block.
// ---------------------------------------------------------------------------

#ifndef __has_builtin
#define __has_builtin(x) 0
#endif
#if defined(__HIP_DEVICE_COMPILE__) && \
    __has_builtin(__builtin_amdgcn_tensor_load_to_lds)
#define USE_TDM 1
#else
#define USE_TDM 0
#endif

typedef __bf16 bf16_t;
typedef bf16_t v16bf __attribute__((ext_vector_type(16)));
typedef float  v8f   __attribute__((ext_vector_type(8)));
typedef float  f32x4 __attribute__((ext_vector_type(4)));
typedef unsigned int u32x4 __attribute__((ext_vector_type(4)));
typedef int i32x4 __attribute__((ext_vector_type(4)));
typedef int i32x8 __attribute__((ext_vector_type(8)));

union Frag {
  u32x4 q[2];   // 2 x 16B loads
  v16bf v;      // 16 bf16 elements (8 VGPRs)
};
union Pack8 {
  u32x4 q;
  unsigned short s[8];
};

__device__ __forceinline__ v8f wmma_bf16(v16bf a, v16bf b, v8f c) {
  // D = A(16x32 bf16) * B(32x16 bf16) + C(16x16 f32)
  return __builtin_amdgcn_wmma_f32_16x16x32_bf16(false, a, false, b,
                                                 (short)0, c, false, false);
}
__device__ __forceinline__ unsigned short f2bf(float f) {
  union { float f; unsigned int u; } t; t.f = f;
  unsigned int u = t.u;
  return (unsigned short)((u + 0x7fffu + ((u >> 16) & 1u)) >> 16); // RNE
}
__device__ __forceinline__ float bf2f(unsigned short h) {
  union { unsigned int u; float f; } t; t.u = ((unsigned int)h) << 16;
  return t.f;
}
__device__ __forceinline__ v8f vzero8() {
  v8f z = {0.f,0.f,0.f,0.f,0.f,0.f,0.f,0.f}; return z;
}
// A-matrix 16x32 bf16 fragment (row baked into p):
// elems 0..7 -> K=8*hi+i, elems 8..15 -> K=16+8*hi+i
template <typename P>
__device__ __forceinline__ Frag load_afrag(P p, int hi) {
  Frag f;
  f.q[0] = *(const u32x4*)(p + 8*hi);
  f.q[1] = *(const u32x4*)(p + 16 + 8*hi);
  return f;
}
// B-matrix 32x16 bf16 fragment (col baked into p):
// elems 0..7 -> K=8*hi+i, elems 8..15 -> K=16+8*hi+i
template <typename P>
__device__ __forceinline__ Frag load_bfrag(P p, int hi) {
  Frag f;
  f.q[0] = *(const u32x4*)(p + 8*hi);
  f.q[1] = *(const u32x4*)(p + 16 + 8*hi);
  return f;
}

#if USE_TDM
// ---------------------------------------------------------------------------
// Tensor Data Mover: load a 2-D bf16 tile [rows x cols] (row stride in elems)
// from global into LDS.  Descriptor per CDNA5 ISA ch.8 (D# groups 0/1).
// ---------------------------------------------------------------------------
__device__ __forceinline__ void tdm_load_tile_2d(unsigned lds_off,
                                                 const void* gptr, int rows,
                                                 int cols, int stride_elems) {
  unsigned long long ga = (unsigned long long)gptr;
  u32x4 g0;
  g0[0] = 1u;                                   // count=1 (user descriptor)
  g0[1] = lds_off;                              // lds_addr (bytes)
  g0[2] = (unsigned)(ga & 0xffffffffull);       // global_addr[31:0]
  g0[3] = (unsigned)((ga >> 32) & 0x1ffffffull) // global_addr[56:32]
          | (2u << 30);                         // type=2 ("image")
  i32x8 g1;
  g1[0] = (1 << 16);                            // data_size=1 (2 bytes)
  g1[1] = (int)((cols & 0xffff) << 16);         // tensor_dim0[15:0]
  g1[2] = (int)(((unsigned)cols >> 16) | ((rows & 0xffff) << 16));
  g1[3] = (int)(((unsigned)rows >> 16) | ((cols & 0xffff) << 16)); // tile_dim0
  g1[4] = rows & 0xffff;                        // tile_dim1 (tile_dim2=0)
  g1[5] = stride_elems;                         // tensor_dim0_stride[31:0]
  g1[6] = 0;
  g1[7] = 0;
  i32x4 z4 = {0, 0, 0, 0};
#if __clang_major__ >= 23
  i32x8 z8 = {0, 0, 0, 0, 0, 0, 0, 0};
  __builtin_amdgcn_tensor_load_to_lds(g0, g1, z4, z4, z8, 0);
#else
  __builtin_amdgcn_tensor_load_to_lds(g0, g1, z4, z4, 0);
#endif
}
#endif

// ---------------------------------------------------------------------------
// f32 -> bf16 conversion, 8 elems / thread (b128 in, b128 out)
// ---------------------------------------------------------------------------
__global__ void cvt_kernel(const float* __restrict__ src,
                           unsigned short* __restrict__ dst, long n8) {
  long i = (long)blockIdx.x * blockDim.x + threadIdx.x;
  if (i >= n8) return;
  const f32x4* s4 = (const f32x4*)src;
  f32x4 a = s4[2 * i], b = s4[2 * i + 1];
  Pack8 o;
#pragma unroll
  for (int j = 0; j < 4; ++j) { o.s[j] = f2bf(a[j]); o.s[4 + j] = f2bf(b[j]); }
  ((u32x4*)dst)[i] = o.q;
}

// f32 [K,N] -> bf16 transposed [N,K] (so GEMM B-fragments are contiguous)
__global__ void cvt_t_kernel(const float* __restrict__ src,
                             unsigned short* __restrict__ dst, int K, int N) {
  long i = (long)blockIdx.x * blockDim.x + threadIdx.x;
  if (i >= (long)K * N) return;
  int n = (int)(i / K), k = (int)(i % K);
  dst[i] = f2bf(src[(size_t)k * N + n]);
}

// ---------------------------------------------------------------------------
// bf16 GEMM: out = A[M,K] @ W[K,N] (+bias)(+resid)(gelu?), W given as Wt[N,K].
// block = 256 threads (8 waves); wave tile 16M x 64N.  No LDS, no barriers:
// B-fragments stream from the L2-resident transposed weights via b128 loads.
// ---------------------------------------------------------------------------
__global__ __launch_bounds__(256)
void gemm_bf16_kernel(const unsigned short* __restrict__ A,
                      const unsigned short* __restrict__ Wt,
                      const float* __restrict__ bias,
                      const float* __restrict__ resid,
                      float* __restrict__ outF,
                      unsigned short* __restrict__ outB,
                      int M, int N, int K, int dogelu) {
  const int tid  = threadIdx.x;
  const int wave = tid >> 5, lane = tid & 31;
  const int m = lane & 15, hi = lane >> 4;
  const int rowA    = blockIdx.x * 128 + wave * 16 + m;
  const int colBase = blockIdx.y * 64;

  const unsigned short* arow = A + (size_t)rowA * K;
  const unsigned short* bp0 = Wt + (size_t)(colBase +  0 + m) * K;
  const unsigned short* bp1 = Wt + (size_t)(colBase + 16 + m) * K;
  const unsigned short* bp2 = Wt + (size_t)(colBase + 32 + m) * K;
  const unsigned short* bp3 = Wt + (size_t)(colBase + 48 + m) * K;

  v8f acc[4];
  acc[0]=vzero8(); acc[1]=vzero8(); acc[2]=vzero8(); acc[3]=vzero8();

  for (int k0 = 0; k0 < K; k0 += 64) {
    __builtin_prefetch(arow + k0 + 64, 0, 1);   // global_prefetch_b8
#pragma unroll
    for (int u = 0; u < 2; ++u) {
      const int kk = k0 + u * 32;
      Frag af = load_afrag(arow + kk, hi);
      Frag f0 = load_bfrag(bp0 + kk, hi);
      Frag f1 = load_bfrag(bp1 + kk, hi);
      Frag f2 = load_bfrag(bp2 + kk, hi);
      Frag f3 = load_bfrag(bp3 + kk, hi);
      acc[0] = wmma_bf16(af.v, f0.v, acc[0]);
      acc[1] = wmma_bf16(af.v, f1.v, acc[1]);
      acc[2] = wmma_bf16(af.v, f2.v, acc[2]);
      acc[3] = wmma_bf16(af.v, f3.v, acc[3]);
    }
  }

  const int rowC = blockIdx.x * 128 + wave * 16 + 8 * hi;
#pragma unroll
  for (int c = 0; c < 4; ++c) {
    int col = colBase + c * 16 + m;
    float bv = bias ? bias[col] : 0.f;
#pragma unroll
    for (int i = 0; i < 8; ++i) {
      size_t idx = (size_t)(rowC + i) * N + col;
      float vv = acc[c][i] + bv;
      if (resid)  vv += resid[idx];
      if (dogelu) vv = 0.5f * vv * (1.f + erff(vv * 0.70710678118f));
      if (outF) outF[idx] = vv;
      if (outB) outB[idx] = f2bf(vv);
    }
  }
}

// ---------------------------------------------------------------------------
// K1 = k + 0.5*(bhk + pk[s]),  Q2 = 0.5*(bhq + pq[s])  (8 elems / thread)
// ---------------------------------------------------------------------------
__global__ void combine_kernel(const unsigned short* __restrict__ kb,
                               const unsigned short* __restrict__ bhkb,
                               const unsigned short* __restrict__ pkb,
                               const unsigned short* __restrict__ bhqb,
                               const unsigned short* __restrict__ pqb,
                               unsigned short* __restrict__ K1b,
                               unsigned short* __restrict__ Q2b,
                               int S, int H, long n8) {
  long i = (long)blockIdx.x * blockDim.x + threadIdx.x;
  if (i >= n8) return;
  long e0 = i * 8;
  int col = (int)(e0 % H);
  int s   = (int)((e0 / H) % S);
  long pi = (long)s * H + col;
  Pack8 kk, bk, pk, bq, pq, o1, o2;
  kk.q = *(const u32x4*)(kb + e0);
  bk.q = *(const u32x4*)(bhkb + e0);
  pk.q = *(const u32x4*)(pkb + pi);
  bq.q = *(const u32x4*)(bhqb + e0);
  pq.q = *(const u32x4*)(pqb + pi);
#pragma unroll
  for (int j = 0; j < 8; ++j) {
    o1.s[j] = f2bf(bf2f(kk.s[j]) + 0.5f * (bf2f(bk.s[j]) + bf2f(pk.s[j])));
    o2.s[j] = f2bf(0.5f * (bf2f(bq.s[j]) + bf2f(pq.s[j])));
  }
  *(u32x4*)(K1b + e0) = o1.q;
  *(u32x4*)(Q2b + e0) = o2.q;
}

// ---------------------------------------------------------------------------
// Flash attention: scores = (q.K1^T + Q2.k^T)/8 + mask; online softmax; O=P.V
// block = 256 threads (8 waves) handles one (b,h) and 128 query rows.
// K1/k 32x64 bf16 tiles -> LDS via TDM (tensor_load_to_lds) where available,
// V staged transposed.  12 WMMAs / key tile / wave.
// ---------------------------------------------------------------------------
__global__ __launch_bounds__(256)
void attn_kernel(const unsigned short* __restrict__ Qb,
                 const unsigned short* __restrict__ K1b,
                 const unsigned short* __restrict__ Kb,
                 const unsigned short* __restrict__ Q2b,
                 const unsigned short* __restrict__ Vb,
                 const int* __restrict__ iseq,
                 unsigned short* __restrict__ Ctx, int S) {
  const int H = 512;
  const int b = blockIdx.x >> 3;
  const int h = blockIdx.x & 7;
  const int tid = threadIdx.x, wave = tid >> 5, lane = tid & 31;
  const int m = lane & 15, hi = lane >> 4;
  const int qrowA = blockIdx.y * 128 + wave * 16 + m;   // A-frag row
  const int qrow0 = blockIdx.y * 128 + wave * 16;       // wave tile base

  __shared__ unsigned short K1s[32 * 64];
  __shared__ unsigned short Ks [32 * 64];
  __shared__ unsigned short Vt [64 * 32];               // [d][key]
  __shared__ float validf[32];
  __shared__ unsigned short Ps[8][16 * 32];             // per-wave P staging

  size_t qbase = ((size_t)(b * S + qrowA)) * H + h * 64;
  Frag qf0  = load_afrag(Qb  + qbase,      hi);         // dims 0..31
  Frag qf1  = load_afrag(Qb  + qbase + 32, hi);         // dims 32..63
  Frag q2f0 = load_afrag(Q2b + qbase,      hi);
  Frag q2f1 = load_afrag(Q2b + qbase + 32, hi);

  v8f o[4];
  o[0]=vzero8(); o[1]=vzero8(); o[2]=vzero8(); o[3]=vzero8();
  float mrow[8], lrow[8];
#pragma unroll
  for (int i = 0; i < 8; ++i) { mrow[i] = -3.0e38f; lrow[i] = 0.f; }

  const int keyS = tid >> 3, d0 = (tid & 7) * 8;        // staging coords

  for (int t = 0; t < S / 32; ++t) {
    const int kb0 = t * 32;
    __syncthreads();                                    // tiles free to reuse
    {
      size_t gbase = ((size_t)(b * S + kb0)) * H + h * 64;
#if USE_TDM
      if (wave == 0) {                                  // wave-uniform issue
        tdm_load_tile_2d((unsigned)(unsigned long long)(void*)K1s,
                         K1b + gbase, 32, 64, H);
        tdm_load_tile_2d((unsigned)(unsigned long long)(void*)Ks,
                         Kb + gbase, 32, 64, H);
        __builtin_amdgcn_s_wait_tensorcnt(0);
      }
#else
      size_t g = gbase + (size_t)keyS * H + d0;         // 1 thread : 8 elems
      *(u32x4*)&K1s[keyS * 64 + d0] = *(const u32x4*)&K1b[g];
      *(u32x4*)&Ks [keyS * 64 + d0] = *(const u32x4*)&Kb[g];
#endif
      // V tile, transposed for B-fragments: one b128 load, 8 b16 LDS stores
      Pack8 vv;
      vv.q = *(const u32x4*)&Vb[gbase + (size_t)keyS * H + d0];
#pragma unroll
      for (int j = 0; j < 8; ++j) Vt[(d0 + j) * 32 + keyS] = vv.s[j];
      if (tid < 32) validf[tid] = (iseq[b * S + kb0 + tid] > 0) ? 1.f : 0.f;
    }
    __syncthreads();

    // ---- scores: 16x32 tile as two 16x16 C fragments, 8 WMMAs
    v8f s[2]; s[0] = vzero8(); s[1] = vzero8();
#pragma unroll
    for (int c = 0; c < 2; ++c) {
      Frag b10 = load_bfrag(&K1s[(c * 16 + m) * 64 +  0], hi);
      Frag b11 = load_bfrag(&K1s[(c * 16 + m) * 64 + 32], hi);
      Frag bk0 = load_bfrag(&Ks [(c * 16 + m) * 64 +  0], hi);
      Frag bk1 = load_bfrag(&Ks [(c * 16 + m) * 64 + 32], hi);
      s[c] = wmma_bf16(qf0.v,  b10.v, s[c]);
      s[c] = wmma_bf16(qf1.v,  b11.v, s[c]);
      s[c] = wmma_bf16(q2f0.v, bk0.v, s[c]);
      s[c] = wmma_bf16(q2f1.v, bk1.v, s[c]);
    }

    // ---- mask + online softmax (rows live in 16-lane groups)
    float corr[8];
#pragma unroll
    for (int i = 0; i < 8; ++i) {
      int row = qrow0 + 8 * hi + i;
      int j0 = kb0 + m, j1 = kb0 + 16 + m;
      float p0 = s[0][i] * 0.125f +
                 (((j0 <= row) && (validf[m] > 0.f)) ? 0.f : -10000.f);
      float p1 = s[1][i] * 0.125f +
                 (((j1 <= row) && (validf[16 + m] > 0.f)) ? 0.f : -10000.f);
      float mx = fmaxf(p0, p1);
      mx = fmaxf(mx, __shfl_xor(mx, 1));
      mx = fmaxf(mx, __shfl_xor(mx, 2));
      mx = fmaxf(mx, __shfl_xor(mx, 4));
      mx = fmaxf(mx, __shfl_xor(mx, 8));
      float mnew = fmaxf(mrow[i], mx);
      float cr = __expf(mrow[i] - mnew);
      float e0 = __expf(p0 - mnew), e1 = __expf(p1 - mnew);
      float rs = e0 + e1;
      rs += __shfl_xor(rs, 1);
      rs += __shfl_xor(rs, 2);
      rs += __shfl_xor(rs, 4);
      rs += __shfl_xor(rs, 8);
      lrow[i] = lrow[i] * cr + rs;
      mrow[i] = mnew;
      corr[i] = cr;
      Ps[wave][(8 * hi + i) * 32 + m]      = f2bf(e0);  // C-layout -> LDS
      Ps[wave][(8 * hi + i) * 32 + 16 + m] = f2bf(e1);
    }
#pragma unroll
    for (int i = 0; i < 8; ++i) {
      float cr = corr[i];
      o[0][i] *= cr; o[1][i] *= cr; o[2][i] *= cr; o[3][i] *= cr;
    }
    asm volatile("" ::: "memory");  // keep same-wave LDS stores before loads

    // ---- O += P(16x32) @ V(32x64): 4 WMMAs
    Frag pf = load_afrag(&Ps[wave][m * 32], hi);        // P in A-layout
#pragma unroll
    for (int c = 0; c < 4; ++c) {
      Frag vf = load_bfrag(&Vt[(c * 16 + m) * 32], hi);
      o[c] = wmma_bf16(pf.v, vf.v, o[c]);
    }
  }

#pragma unroll
  for (int c = 0; c < 4; ++c)
#pragma unroll
    for (int i = 0; i < 8; ++i) {
      int row = qrow0 + 8 * hi + i;
      int d = h * 64 + c * 16 + m;
      Ctx[((size_t)(b * S + row)) * H + d] = f2bf(o[c][i] / lrow[i]);
    }
}

// ---------------------------------------------------------------------------
// LayerNorm over H=512, one block (128 threads) per row, float4 I/O
// ---------------------------------------------------------------------------
__global__ __launch_bounds__(128)
void ln_kernel(const float* __restrict__ in, const float* __restrict__ g,
               const float* __restrict__ bta, float* __restrict__ outF,
               unsigned short* __restrict__ outB) {
  const int row = blockIdx.x, tid = threadIdx.x;
  const f32x4* r4 = (const f32x4*)(in + (size_t)row * 512);
  f32x4 x = r4[tid];
  float s = 0.f, s2 = 0.f;
#pragma unroll
  for (int i = 0; i < 4; ++i) { s += x[i]; s2 += x[i] * x[i]; }
  __shared__ float rs[128], rq[128];
  rs[tid] = s; rq[tid] = s2;
  __syncthreads();
  for (int off = 64; off > 0; off >>= 1) {
    if (tid < off) { rs[tid] += rs[tid + off]; rq[tid] += rq[tid + off]; }
    __syncthreads();
  }
  float mean = rs[0] * (1.f / 512.f);
  float var  = rq[0] * (1.f / 512.f) - mean * mean;
  float rstd = rsqrtf(var + 1e-12f);
#pragma unroll
  for (int i = 0; i < 4; ++i) {
    int c = tid * 4 + i;
    float y = (x[i] - mean) * rstd * g[c] + bta[c];
    if (outF) outF[(size_t)row * 512 + c] = y;
    if (outB) outB[(size_t)row * 512 + c] = f2bf(y);
  }
}

// ---------------------------------------------------------------------------
// Host orchestration
// ---------------------------------------------------------------------------
extern "C" void kernel_launch(void* const* d_in, const int* in_sizes, int n_in,
                              void* d_out, int out_size, void* d_ws,
                              size_t ws_size, hipStream_t stream) {
  (void)in_sizes; (void)n_in; (void)out_size; (void)ws_size;
  const int B = 32, S = 512, H = 512, INNER = 2048;
  const int NTOK = B * S;                 // 16384
  const long NE = (long)NTOK * H;         // 8388608

  const float* x    = (const float*)d_in[0];
  const float* vin  = (const float*)d_in[1];
  const float* beh  = (const float*)d_in[2];
  const float* pos  = (const float*)d_in[3];
  const float* Wq = (const float*)d_in[4];   const float* bq = (const float*)d_in[5];
  const float* Wk = (const float*)d_in[6];   const float* bk = (const float*)d_in[7];
  const float* Wv = (const float*)d_in[8];   const float* bv = (const float*)d_in[9];
  const float* Wpk = (const float*)d_in[10]; const float* bpk = (const float*)d_in[11];
  const float* Wpq = (const float*)d_in[12]; const float* bpq = (const float*)d_in[13];
  const float* Wbk = (const float*)d_in[14]; const float* bbk = (const float*)d_in[15];
  const float* Wbq = (const float*)d_in[16]; const float* bbq = (const float*)d_in[17];
  const float* Wd = (const float*)d_in[18];  const float* bd = (const float*)d_in[19];
  const float* ln_g = (const float*)d_in[20]; const float* ln_b = (const float*)d_in[21];
  const float* W1 = (const float*)d_in[22];  const float* b1 = (const float*)d_in[23];
  const float* W2 = (const float*)d_in[24];  const float* b2 = (const float*)d_in[25];
  const float* ln2_g = (const float*)d_in[26]; const float* ln2_b = (const float*)d_in[27];
  const int* iseq = (const int*)d_in[28];
  float* out = (float*)d_out;

  char* wsp = (char*)d_ws;
  size_t off = 0;
  auto alloc = [&](size_t bytes) -> void* {
    void* p = wsp + off;
    off += (bytes + 255) & ~(size_t)255;
    return p;
  };
  typedef unsigned short u16;
  u16* xb    = (u16*)alloc(NE * 2);
  u16* vib   = (u16*)alloc(NE * 2);
  u16* behab = (u16*)alloc(NE * 2);
  u16* posb  = (u16*)alloc((size_t)S * H * 2);
  u16* Wqt  = (u16*)alloc((size_t)H * H * 2);      // all weights transposed
  u16* Wkt  = (u16*)alloc((size_t)H * H * 2);
  u16* Wvt  = (u16*)alloc((size_t)H * H * 2);
  u16* Wpkt = (u16*)alloc((size_t)H * H * 2);
  u16* Wpqt = (u16*)alloc((size_t)H * H * 2);
  u16* Wbkt = (u16*)alloc((size_t)H * H * 2);
  u16* Wbqt = (u16*)alloc((size_t)H * H * 2);
  u16* Wdt  = (u16*)alloc((size_t)H * H * 2);
  u16* W1t  = (u16*)alloc((size_t)H * INNER * 2);
  u16* W2t  = (u16*)alloc((size_t)INNER * H * 2);
  u16* qb   = (u16*)alloc(NE * 2);
  u16* kbuf = (u16*)alloc(NE * 2);
  u16* vbuf = (u16*)alloc(NE * 2);
  u16* bhkb = (u16*)alloc(NE * 2);
  u16* bhqb = (u16*)alloc(NE * 2);
  u16* pkb  = (u16*)alloc((size_t)S * H * 2);
  u16* pqb  = (u16*)alloc((size_t)S * H * 2);
  u16* K1b  = (u16*)alloc(NE * 2);
  u16* Q2b  = (u16*)alloc(NE * 2);
  u16* ctxb = (u16*)alloc(NE * 2);
  float* attn_pre = (float*)alloc(NE * 4);   // reused as ffn_pre
  float* attn_out = (float*)alloc(NE * 4);
  u16* attnb = (u16*)alloc(NE * 2);
  u16* hb    = (u16*)alloc((size_t)NTOK * INNER * 2);

  auto cvt = [&](const float* s, u16* d, long n) {
    long n8 = n / 8;
    cvt_kernel<<<(int)((n8 + 255) / 256), 256, 0, stream>>>(s, d, n8);
  };
  auto cvtT = [&](const float* s, u16* d, int K, int N) {
    long n = (long)K * N;
    cvt_t_kernel<<<(int)((n + 255) / 256), 256, 0, stream>>>(s, d, K, N);
  };
  auto gemm = [&](const u16* A, const u16* Wt, const float* bias,
                  const float* resid, float* oF, u16* oB, int M, int N, int K,
                  int gl) {
    dim3 g(M / 128, N / 64);
    gemm_bf16_kernel<<<g, 256, 0, stream>>>(A, Wt, bias, resid, oF, oB, M, N,
                                            K, gl);
  };

  // 1) bf16 conversions; weights transposed once -> contiguous B-fragments,
  //    resident in 192MB L2 for the rest of the pipeline.
  cvt(x, xb, NE); cvt(vin, vib, NE); cvt(beh, behab, NE);
  cvt(pos, posb, (long)S * H);
  cvtT(Wq, Wqt, H, H);   cvtT(Wk, Wkt, H, H);
  cvtT(Wv, Wvt, H, H);   cvtT(Wpk, Wpkt, H, H);
  cvtT(Wpq, Wpqt, H, H); cvtT(Wbk, Wbkt, H, H);
  cvtT(Wbq, Wbqt, H, H); cvtT(Wd, Wdt, H, H);
  cvtT(W1, W1t, H, INNER); cvtT(W2, W2t, INNER, H);

  // 2) projections
  gemm(posb, Wpkt, bpk, nullptr, nullptr, pkb, S, H, H, 0);
  gemm(posb, Wpqt, bpq, nullptr, nullptr, pqb, S, H, H, 0);
  gemm(xb,   Wqt,  bq,  nullptr, nullptr, qb,   NTOK, H, H, 0);
  gemm(xb,   Wkt,  bk,  nullptr, nullptr, kbuf, NTOK, H, H, 0);
  gemm(vib,  Wvt,  bv,  vin,     nullptr, vbuf, NTOK, H, H, 0); // +value resid
  gemm(behab, Wbkt, bbk, nullptr, nullptr, bhkb, NTOK, H, H, 0);
  gemm(behab, Wbqt, bbq, nullptr, nullptr, bhqb, NTOK, H, H, 0);

  // 3) fold score terms: K1 = k + .5(bhk+pk), Q2 = .5(bhq+pq)
  combine_kernel<<<(int)((NE / 8 + 255) / 256), 256, 0, stream>>>(
      kbuf, bhkb, pkb, bhqb, pqb, K1b, Q2b, S, H, NE / 8);

  // 4) flash attention
  attn_kernel<<<dim3(B * 8, S / 128), 256, 0, stream>>>(qb, K1b, kbuf, Q2b,
                                                        vbuf, iseq, ctxb, S);

  // 5) output projection + residual, LN1
  gemm(ctxb, Wdt, bd, x, attn_pre, nullptr, NTOK, H, H, 0);
  ln_kernel<<<NTOK, 128, 0, stream>>>(attn_pre, ln_g, ln_b, attn_out, attnb);

  // 6) FFN (exact gelu) + residual, LN2 -> d_out
  gemm(attnb, W1t, b1, nullptr, nullptr, hb, NTOK, INNER, H, 1);
  gemm(hb, W2t, b2, attn_out, attn_pre, nullptr, NTOK, H, INNER, 0);
  ln_kernel<<<NTOK, 128, 0, stream>>>(attn_pre, ln2_g, ln2_b, out, nullptr);
}
